// SparseAutoencoder_7473243095085
// MI455X (gfx1250) — matrix operand order repaired and checked by reading, not compiled, not run on hardware
//
#include <hip/hip_runtime.h>
#include <hip/hip_bf16.h>

// ---------------------------------------------------------------------------
// Types for CDNA5 WMMA (wave32)
// ---------------------------------------------------------------------------
typedef __bf16 v16bf __attribute__((ext_vector_type(16)));
typedef float  v8f   __attribute__((ext_vector_type(8)));

union FragBF {
    v16bf v;
    uint4 q[2];   // 32 bytes = 16 bf16
};

__device__ __forceinline__ unsigned short f2bf(float f) {
    unsigned int u = __float_as_uint(f);
    u += 0x7FFFu + ((u >> 16) & 1u);   // round-to-nearest-even
    return (unsigned short)(u >> 16);
}

// Pack two f32 into two bf16 with round-half-up: one v_add per element plus a
// single v_perm_b32 selecting the two high halves. 1.5 VALU/elem -> hides
// entirely under WMMA co-execution.
__device__ __forceinline__ unsigned cvt_pk_bf16(float x, float y) {
#if __has_builtin(__builtin_amdgcn_perm)
    unsigned ux = __float_as_uint(x) + 0x8000u;
    unsigned uy = __float_as_uint(y) + 0x8000u;
    // result bytes {uy[3],uy[2],ux[3],ux[2]} : sel 7,6 from src0(=uy), 3,2 from src1(=ux)
    return __builtin_amdgcn_perm(uy, ux, 0x07060302u);
#else
    return (unsigned)f2bf(x) | ((unsigned)f2bf(y) << 16);
#endif
}

// ---------------------------------------------------------------------------
// Tiled bf16 WMMA GEMM:  C[M,N] = act( A[M,K] * B[N,K]^T + bias[N] )
// Both A and B are row-major with K contiguous (fits x@W_enc^T and z@W_dec^T).
// Block tile 128x128, BK=64, 256 threads (8 waves), wave tile 32x64.
// Software-pipelined: next K-tile is fetched into registers while the current
// tile's WMMAs run from LDS.
// ---------------------------------------------------------------------------
#define BM 128
#define BN 128
#define BK 64
#define LDS_STRIDE 72   // 64 + 8 pad (bf16 elements) -> 144B row, kills bank conflicts

template<bool RELU>
__global__ __launch_bounds__(256)
void gemm_bf16_wmma(const float* __restrict__ A, const float* __restrict__ Bm,
                    const float* __restrict__ bias, float* __restrict__ C,
                    int K, int lda, int ldb, int ldc)
{
    __shared__ alignas(16) unsigned short sA[BM * LDS_STRIDE];
    __shared__ alignas(16) unsigned short sB[BN * LDS_STRIDE];

    const int tid  = threadIdx.x;
    const int wave = tid >> 5;       // 8 waves (wave32)
    const int lane = tid & 31;
    const int half = lane >> 4;      // 0: lanes 0-15, 1: lanes 16-31
    const int l15  = lane & 15;

    const int m0 = blockIdx.x * BM;  // M fastest-varying: co-resident blocks share B panel
    const int n0 = blockIdx.y * BN;
    const int wm = (wave & 3) * 32;  // wave M offset: 2 x 16 tiles
    const int wn = (wave >> 2) * 64; // wave N offset: 4 x 16 tiles

    // Per-thread staging coordinates: 128 rows x 16 float4 columns per tile.
    const int srow = tid >> 4;       // 0..15 base row (stride 16 per step)
    const int scol = tid & 15;       // float4 column 0..15

    const float* gA = A  + (size_t)(m0 + srow) * lda + scol * 4;
    const float* gB = Bm + (size_t)(n0 + srow) * ldb + scol * 4;

    v8f acc[2][4];
    const v8f vzero = {0.f, 0.f, 0.f, 0.f, 0.f, 0.f, 0.f, 0.f};
#pragma unroll
    for (int i = 0; i < 2; ++i)
#pragma unroll
        for (int j = 0; j < 4; ++j) acc[i][j] = vzero;

    float4 rA[8], rB[8];
    // Preload tile kk = 0
#pragma unroll
    for (int r = 0; r < 8; ++r) {
        rA[r] = *(const float4*)(gA + (size_t)r * 16 * lda);
        rB[r] = *(const float4*)(gB + (size_t)r * 16 * ldb);
    }

    for (int kk = 0; kk < K; kk += BK) {
        __syncthreads();   // previous tile's fragments consumed before overwrite

        // Convert + store staged tile to LDS (1 add/elem + 1 perm/pair)
#pragma unroll
        for (int r = 0; r < 8; ++r) {
            const int row = srow + r * 16;
            uint2 pa;
            pa.x = cvt_pk_bf16(rA[r].x, rA[r].y);
            pa.y = cvt_pk_bf16(rA[r].z, rA[r].w);
            *(uint2*)(&sA[row * LDS_STRIDE + scol * 4]) = pa;
            uint2 pb;
            pb.x = cvt_pk_bf16(rB[r].x, rB[r].y);
            pb.y = cvt_pk_bf16(rB[r].z, rB[r].w);
            *(uint2*)(&sB[row * LDS_STRIDE + scol * 4]) = pb;
        }
        __syncthreads();

        // Kick off next tile's global loads; they overlap the WMMAs below.
        if (kk + BK < K) {
            const float* nA = gA + (kk + BK);
            const float* nB = gB + (kk + BK);
#pragma unroll
            for (int r = 0; r < 8; ++r) {
                rA[r] = *(const float4*)(nA + (size_t)r * 16 * lda);
                rB[r] = *(const float4*)(nB + (size_t)r * 16 * ldb);
            }
        }

#pragma unroll
        for (int ks = 0; ks < BK; ks += 32) {
            FragBF a[2], b[4];
            // A 16x32 bf16 layout: lane<16 -> row M=l15, K = ks+{0..7} and ks+{16..23};
            // lane>=16 -> same row, K = ks+8+{0..7} and ks+24+{0..7}.
#pragma unroll
            for (int i = 0; i < 2; ++i) {
                const uint4* pa = (const uint4*)
                    &sA[(wm + i * 16 + l15) * LDS_STRIDE + ks + half * 8];
                a[i].q[0] = pa[0];        // K run +0..7
                a[i].q[1] = pa[2];        // K run +16..23 (32B further)
            }
            // B 32x16 bf16 layout: lane -> column N=l15, K = ks+half*8+{0..7} and ks+16+half*8+{0..7}
#pragma unroll
            for (int j = 0; j < 4; ++j) {
                const uint4* pb = (const uint4*)
                    &sB[(wn + j * 16 + l15) * LDS_STRIDE + ks + half * 8];
                b[j].q[0] = pb[0];
                b[j].q[1] = pb[2];
            }
#pragma unroll
            for (int i = 0; i < 2; ++i)
#pragma unroll
                for (int j = 0; j < 4; ++j)
                    acc[i][j] = __builtin_amdgcn_wmma_f32_16x16x32_bf16(
                        false, a[i].v, false, b[j].v,
                        (short)0, acc[i][j], false, false);
        }
    }

    // Epilogue: bias (+relu), scatter per C-tile ISA layout:
    // VGPR r holds (M = r + half*8, N = l15) of each 16x16 tile.
    float bv[4];
#pragma unroll
    for (int j = 0; j < 4; ++j) bv[j] = bias[n0 + wn + j * 16 + l15];

#pragma unroll
    for (int i = 0; i < 2; ++i) {
#pragma unroll
        for (int r = 0; r < 8; ++r) {
            const int row = m0 + wm + i * 16 + half * 8 + r;
            float* cp = C + (size_t)row * ldc + n0 + wn + l15;
#pragma unroll
            for (int j = 0; j < 4; ++j) {
                float v = acc[i][j][r] + bv[j];
                if (RELU) v = fmaxf(v, 0.f);
                cp[j * 16] = v;
            }
        }
    }
}

// ---------------------------------------------------------------------------
// Exact top-K (K=64) mask per row, in place. Values are >= 0 (post-relu) so
// the f32 bit pattern is monotonic as uint. Two-level 13-bit radix select
// (26-bit threshold prefix) + tie budget => exactly K survivors.
// ---------------------------------------------------------------------------
#define TK_T  256
#define NBINS 8192

__global__ __launch_bounds__(TK_T)
void topk_mask_kernel(float* __restrict__ z, int H, int K)
{
    __shared__ int hist[NBINS];
    __shared__ int csum[TK_T];
    __shared__ int s_p0, s_p1, s_need, s_taken;

    const int tid = threadIdx.x;
    float* row = z + (size_t)blockIdx.x * H;

    // ---- level 0: histogram of bits[30:18] ----
    for (int b = tid; b < NBINS; b += TK_T) hist[b] = 0;
    __syncthreads();
    for (int i = tid; i < H; i += TK_T) {
        unsigned u = __float_as_uint(row[i]);
        atomicAdd(&hist[u >> 18], 1);
    }
    __syncthreads();
    {
        int s = 0;
#pragma unroll 8
        for (int j = 0; j < 32; ++j) s += hist[tid * 32 + j];
        csum[tid] = s;
    }
    __syncthreads();
    if (tid == 0) {
        int target = K, c = 0, cc = 255;
        while (cc > 0 && c + csum[cc] < target) { c += csum[cc]; --cc; }
        int b = cc * 32 + 31;
        while (b > cc * 32 && c + hist[b] < target) { c += hist[b]; --b; }
        s_p0 = b;
        s_need = target - c;     // how many to take from bin b (>=1)
    }
    __syncthreads();
    const int p0 = s_p0;

    // ---- level 1: among values with top bin == p0, histogram bits[17:5] ----
    for (int b = tid; b < NBINS; b += TK_T) hist[b] = 0;
    __syncthreads();
    for (int i = tid; i < H; i += TK_T) {
        unsigned u = __float_as_uint(row[i]);
        if ((int)(u >> 18) == p0) atomicAdd(&hist[(u >> 5) & 8191], 1);
    }
    __syncthreads();
    {
        int s = 0;
#pragma unroll 8
        for (int j = 0; j < 32; ++j) s += hist[tid * 32 + j];
        csum[tid] = s;
    }
    __syncthreads();
    if (tid == 0) {
        int target = s_need, c = 0, cc = 255;
        while (cc > 0 && c + csum[cc] < target) { c += csum[cc]; --cc; }
        int b = cc * 32 + 31;
        while (b > cc * 32 && c + hist[b] < target) { c += hist[b]; --b; }
        s_p1 = b;
        s_need = target - c;     // ties to accept at 26-bit prefix (>=1)
        s_taken = 0;
    }
    __syncthreads();
    const int p1 = s_p1;

    // ---- mask pass: zero everything below the 26-bit threshold prefix ----
    for (int i = tid; i < H; i += TK_T) {
        unsigned u = __float_as_uint(row[i]);
        int hi = (int)(u >> 18);
        bool keep;
        if (hi > p0)      keep = true;
        else if (hi < p0) keep = false;
        else {
            int mid = (int)((u >> 5) & 8191);
            if (mid > p1)      keep = true;
            else if (mid < p1) keep = false;
            else               keep = (atomicAdd(&s_taken, 1) < s_need);
        }
        if (!keep) row[i] = 0.f;
    }
}

// ---------------------------------------------------------------------------
// Launch: encoder GEMM -> top-k mask (in place on z output) -> decoder GEMM
// d_out = [ x_hat (B*D) | z_masked (B*H) ]  (float32)
// ---------------------------------------------------------------------------
extern "C" void kernel_launch(void* const* d_in, const int* in_sizes, int n_in,
                              void* d_out, int out_size, void* d_ws, size_t ws_size,
                              hipStream_t stream)
{
    (void)in_sizes; (void)n_in; (void)out_size; (void)d_ws; (void)ws_size;

    const float* x     = (const float*)d_in[0];
    const float* W_enc = (const float*)d_in[1];   // [H, D], K-contiguous
    const float* b_enc = (const float*)d_in[2];
    const float* W_dec = (const float*)d_in[3];   // [D, H], K-contiguous
    const float* b_dec = (const float*)d_in[4];

    const int B = 4096, D = 2048, H = 32768, K = 64;

    float* x_hat = (float*)d_out;
    float* z     = x_hat + (size_t)B * D;

    // Encoder: z = relu(x @ W_enc^T + b_enc)   [4096 x 32768]
    gemm_bf16_wmma<true><<<dim3(B / BM, H / BN), dim3(256), 0, stream>>>(
        x, W_enc, b_enc, z, /*K=*/D, /*lda=*/D, /*ldb=*/D, /*ldc=*/H);

    // Top-K mask in place (exactly K survivors per row)
    topk_mask_kernel<<<dim3(B), dim3(TK_T), 0, stream>>>(z, H, K);

    // Decoder: x_hat = z_masked @ W_dec^T + b_dec   [4096 x 2048]
    gemm_bf16_wmma<false><<<dim3(B / BM, D / BN), dim3(256), 0, stream>>>(
        z, W_dec, b_dec, x_hat, /*K=*/H, /*lda=*/H, /*ldb=*/H, /*ldc=*/D);
}
